// _ViT_34213709480696
// MI455X (gfx1250) — matrix operand order repaired and checked by reading, not compiled, not run on hardware
//
#include <hip/hip_runtime.h>
#include <hip/hip_bf16.h>
#include <math.h>

typedef __attribute__((ext_vector_type(16))) _Float16 v16h;
typedef __attribute__((ext_vector_type(8)))  _Float16 v8h;
typedef __attribute__((ext_vector_type(8)))  float    v8f;

// ---------------------------------------------------------------------------
// WMMA fragment loaders for V_WMMA_F32_16X16X32_F16 (wave32), following the
// contiguous k order below (A and B agree; not the ISA's table k order):
//   A (16x32 f16): lane = kg*16 + m; half i -> K = kg*16 + i  (row m)
//   B (32x16 f16): lane = kg*16 + n; half i -> K = kg*16 + i  (column n)
// Source memory is row-major with `ld` halves per row; for B the memory holds
// rows indexed by the *N* coordinate, contiguous in K.
// ---------------------------------------------------------------------------
__device__ __forceinline__ v16h load_a_frag(const _Float16* p, int m0, int k0, int ld) {
  const int lane = threadIdx.x & 31;
  const int kg = lane >> 4, m = lane & 15;
  const _Float16* row = p + (size_t)(m0 + m) * ld + k0;
  v16h a = *(const v16h*)(row + kg * 16);
  return a;
}

__device__ __forceinline__ v16h load_b_frag(const _Float16* p, int n0, int k0, int ld) {
  const int lane = threadIdx.x & 31;
  const int kg = lane >> 4, n = lane & 15;
  return *(const v16h*)(p + (size_t)(n0 + n) * ld + k0 + kg * 16);
}

// ---------------------------------------------------------------------------
// Kernel 0: one-shot f32 -> f16 conversion of x, qkv_w, proj_w.
// ---------------------------------------------------------------------------
__global__ __launch_bounds__(256)
void cvt_kernel(const float* __restrict__ x, const float* __restrict__ qkv_w,
                const float* __restrict__ proj_w,
                _Float16* __restrict__ xh, _Float16* __restrict__ qwh,
                _Float16* __restrict__ pwh, long long nx) {
  const long long NQW = 768LL * 256;
  long long i = ((long long)blockIdx.x * 256 + threadIdx.x) * 8;
  const float* src;
  _Float16* dst;
  if (i < nx)            { src = x + i;                   dst = xh + i; }
  else if (i < nx + NQW) { src = qkv_w + (i - nx);        dst = qwh + (i - nx); }
  else                   { src = proj_w + (i - nx - NQW); dst = pwh + (i - nx - NQW); }
#pragma unroll
  for (int j = 0; j < 8; ++j) dst[j] = (_Float16)src[j];
}

// ---------------------------------------------------------------------------
// Kernel 1: QKV projection (f16 WMMA).  out = x @ qkv_w^T + qkv_b, scattered
// into window/head-major f16 layout: {q,k,v}[(bw*8+head)*64 + nwin][hd].
// Block: 256 threads (8 waves), 128(M) x 64(N) tile, K stepped by 32.
// Wave w owns m-strip [16w,16w+16) and all 4 n-tiles: 4 WMMAs / K-step.
// A tile staged in LDS (pure b128 copy of f16); B fragments read directly
// from the L2-resident f16 weights (contiguous 32B per lane).
// ---------------------------------------------------------------------------
__global__ __launch_bounds__(256)
void qkv_kernel(const _Float16* __restrict__ xh, const _Float16* __restrict__ qwh,
                const float* __restrict__ qkv_b,
                _Float16* __restrict__ qs, _Float16* __restrict__ kk,
                _Float16* __restrict__ vv) {
  __shared__ _Float16 As[128 * 32];     // 8 KiB
  const int nblk = blockIdx.x;          // 0..11  (N = 768)
  const int mblk = blockIdx.y;          // 0..2047
  const int tid  = threadIdx.x;
  const int wid  = tid >> 5;            // wave's 16-row m-strip
  const int mbase = mblk * 128;
  const int nbase = nblk * 64;
  const int sr = tid >> 1;              // staging row (128 rows, 2 thr/row)
  const int sc = (tid & 1) * 16;        // staging col (2 x 16 halves)

  v8f acc[4] = {};
  for (int k0 = 0; k0 < 256; k0 += 32) {
    __syncthreads();
    *(v8h*)&As[sr * 32 + sc] =
        *(const v8h*)&xh[(size_t)(mbase + sr) * 256 + k0 + sc];
    *(v8h*)&As[sr * 32 + sc + 8] =
        *(const v8h*)&xh[(size_t)(mbase + sr) * 256 + k0 + sc + 8];
    if (k0 + 32 < 256)
      __builtin_prefetch(&xh[(size_t)(mbase + sr) * 256 + k0 + 32 + sc], 0, 3);
    __syncthreads();
    v16h a = load_a_frag(As, wid * 16, 0, 32);
#pragma unroll
    for (int nt = 0; nt < 4; ++nt) {
      v16h b = load_b_frag(qwh, nbase + nt * 16, k0, 256);
      acc[nt] = __builtin_amdgcn_wmma_f32_16x16x32_f16(false, a, false, b,
                                                       (short)0, acc[nt], false, false);
    }
  }

  const int lane = tid & 31;
  const int colL = lane & 15;
  const int rowoff = 8 * (lane >> 4);   // D layout: M = r + 8*(lane/16)
#pragma unroll
  for (int nt = 0; nt < 4; ++nt) {
    int gn = nbase + nt * 16 + colL;    // 0..767
    float bias = qkv_b[gn];
    int part = gn >> 8;                 // 0=q 1=k 2=v
    int cch  = gn & 255;
    int head = cch >> 5, d = cch & 31;
#pragma unroll
    for (int r = 0; r < 8; ++r) {
      int t   = mbase + wid * 16 + rowoff + r;     // global token
      int b   = t >> 14;                           // 16384 tokens / image
      int rem = t & 16383;
      int hi  = rem >> 7, wi = rem & 127;
      int bw  = (((b << 4) | (hi >> 3)) << 4) | (wi >> 3);
      int nwin = ((hi & 7) << 3) | (wi & 7);
      size_t dst = ((size_t)(bw * 8 + head) * 64 + nwin) * 32 + d;
      _Float16 h = (_Float16)(acc[nt][r] + bias);  // q stored UNSCALED
      if (part == 0) qs[dst] = h;
      else if (part == 1) kk[dst] = h;
      else vv[dst] = h;
    }
  }
}

// ---------------------------------------------------------------------------
// Kernel 2: fused attention core for one (window, head).
// Block: 128 threads (4 waves); wave w owns query rows [16w, 16w+16).
//   S = scale * q k^T  (WMMA) ; S += decomposed rel-pos bias ; softmax ;
//   O = P V (WMMA, V staged transposed in LDS) -> yw[(bw*64+tok)*256 + ch]
// ---------------------------------------------------------------------------
__global__ __launch_bounds__(128)
void attn_kernel(const _Float16* __restrict__ qs, const _Float16* __restrict__ kk,
                 const _Float16* __restrict__ vv,
                 const float* __restrict__ rel_pos_h, const float* __restrict__ rel_pos_w,
                 _Float16* __restrict__ yw) {
  __shared__ float    Sbuf[64 * 64];    // 16 KiB
  __shared__ _Float16 Pf[64 * 64];      //  8 KiB
  __shared__ _Float16 Vt[32 * 64];      //  4 KiB  Vt[d][key]
  __shared__ float    relh[64 * 8];
  __shared__ float    relw[64 * 8];

  const int tid  = threadIdx.x;
  const int wid  = tid >> 5;
  const int lane = tid & 31;
  const int bw   = blockIdx.x >> 3;
  const int head = blockIdx.x & 7;
  const size_t base = ((size_t)bw * 8 + head) * (64 * 32);
  const _Float16* qb = qs + base;
  const _Float16* kb = kk + base;
  const _Float16* vb = vv + base;
  const float scale = 0.17677669529663687f;   // 32^-0.5

  // stage V transposed for the P@V B-operand
#pragma unroll
  for (int j = 0; j < 16; ++j) {
    int idx = tid * 16 + j;             // 2048 elements
    int key = idx >> 5, d = idx & 31;
    Vt[d * 64 + key] = vb[idx];
  }

  // decomposed rel-pos bias partials (uses UNSCALED q, per reference)
#pragma unroll
  for (int j = 0; j < 8; ++j) {
    int e = tid + j * 128;              // 1024 dot products of length 32
    int tok = (e >> 3) & 63;
    int kc  = e & 7;
    int qc  = (e < 512) ? (tok >> 3) : (tok & 7);
    const float* tab = ((e < 512) ? rel_pos_h : rel_pos_w) + (size_t)(qc - kc + 7) * 32;
    const _Float16* qrow = qb + tok * 32;
    float s = 0.f;
#pragma unroll
    for (int c = 0; c < 32; ++c) s += (float)qrow[c] * tab[c];
    if (e < 512) relh[tok * 8 + kc] = s; else relw[tok * 8 + kc] = s;
  }

  // S strip = scale * q @ k^T (K = hd = 32 -> single WMMA per 16x16 tile)
  const int m0 = wid * 16;
  {
    v16h a = load_a_frag(qb, m0, 0, 32);
#pragma unroll
    for (int nt = 0; nt < 4; ++nt) {
      v16h b = load_b_frag(kb, nt * 16, 0, 32);
      v8f c = {};
      c = __builtin_amdgcn_wmma_f32_16x16x32_f16(false, a, false, b, (short)0, c, false, false);
      int row = m0 + 8 * (lane >> 4);
      int col = nt * 16 + (lane & 15);
#pragma unroll
      for (int r = 0; r < 8; ++r) Sbuf[(row + r) * 64 + col] = c[r] * scale;
    }
  }
  __syncthreads();

  // bias add + row softmax; emit P in f16
  if (tid < 64) {
    const int m = tid;
    float mx = -1e30f;
    for (int n = 0; n < 64; ++n) {
      float s = Sbuf[m * 64 + n] + relh[m * 8 + (n >> 3)] + relw[m * 8 + (n & 7)];
      Sbuf[m * 64 + n] = s;
      mx = fmaxf(mx, s);
    }
    float sum = 0.f;
    for (int n = 0; n < 64; ++n) {
      float e = __expf(Sbuf[m * 64 + n] - mx);
      Sbuf[m * 64 + n] = e;
      sum += e;
    }
    float rinv = 1.0f / sum;
    for (int n = 0; n < 64; ++n)
      Pf[m * 64 + n] = (_Float16)(Sbuf[m * 64 + n] * rinv);
  }
  __syncthreads();

  // O = P @ V  (M=16 strip, N=32 -> 2 tiles, K=64 -> 2 WMMA steps)
#pragma unroll
  for (int nt = 0; nt < 2; ++nt) {
    v8f c = {};
#pragma unroll
    for (int ks2 = 0; ks2 < 2; ++ks2) {
      v16h a = load_a_frag(Pf, m0, ks2 * 32, 64);
      v16h b = load_b_frag(Vt, nt * 16, ks2 * 32, 64);
      c = __builtin_amdgcn_wmma_f32_16x16x32_f16(false, a, false, b, (short)0, c, false, false);
    }
    int row = m0 + 8 * (lane >> 4);
    int col = head * 32 + nt * 16 + (lane & 15);
#pragma unroll
    for (int r = 0; r < 8; ++r) {
      size_t t = (size_t)bw * 64 + row + r;        // window-ordered token
      yw[t * 256 + col] = (_Float16)c[r];
    }
  }
}

// ---------------------------------------------------------------------------
// Kernel 3: output projection + fused window_reverse scatter.
// Fully LDS/barrier-free: A fragments from f16 activations, B fragments from
// the L2-resident f16 proj weights.  128(M) x 64(N) tile; wave w owns
// m-strip [16w,16w+16), 4 WMMAs per K-step.
// ---------------------------------------------------------------------------
__global__ __launch_bounds__(256)
void proj_kernel(const _Float16* __restrict__ yw, const _Float16* __restrict__ pwh,
                 const float* __restrict__ proj_b, float* __restrict__ out) {
  const int nblk = blockIdx.x;          // 0..3 (N = 256)
  const int mblk = blockIdx.y;          // 0..2047
  const int tid  = threadIdx.x;
  const int wid  = tid >> 5;
  const int mbase = mblk * 128;
  const int nbase = nblk * 64;
  const int lane = tid & 31;

  v8f acc[4] = {};
#pragma unroll 2
  for (int k0 = 0; k0 < 256; k0 += 32) {
    v16h a = load_a_frag(yw, mbase + wid * 16, k0, 256);
#pragma unroll
    for (int nt = 0; nt < 4; ++nt) {
      v16h b = load_b_frag(pwh, nbase + nt * 16, k0, 256);
      acc[nt] = __builtin_amdgcn_wmma_f32_16x16x32_f16(false, a, false, b,
                                                       (short)0, acc[nt], false, false);
    }
  }

  const int colL = lane & 15;
  const int rowoff = 8 * (lane >> 4);
#pragma unroll
  for (int nt = 0; nt < 4; ++nt) {
    int gn = nbase + nt * 16 + colL;
    float bias = proj_b[gn];
#pragma unroll
    for (int r = 0; r < 8; ++r) {
      int tw = mbase + wid * 16 + rowoff + r;      // window-ordered token
      int bwi = tw >> 6;
      int nwin = tw & 63;
      int b  = bwi >> 8;                           // 16x16 windows per image
      int wy = (bwi >> 4) & 15;
      int wx = bwi & 15;
      int hi = wy * 8 + (nwin >> 3);
      int wi = wx * 8 + (nwin & 7);
      size_t t = ((size_t)b << 14) + (size_t)hi * 128 + wi;
      out[t * 256 + gn] = acc[nt][r] + bias;
    }
  }
}

// ---------------------------------------------------------------------------
extern "C" void kernel_launch(void* const* d_in, const int* in_sizes, int n_in,
                              void* d_out, int out_size, void* d_ws, size_t ws_size,
                              hipStream_t stream) {
  const float* x         = (const float*)d_in[0];
  const float* qkv_w     = (const float*)d_in[1];
  const float* qkv_b     = (const float*)d_in[2];
  const float* proj_w    = (const float*)d_in[3];
  const float* proj_b    = (const float*)d_in[4];
  const float* rel_pos_h = (const float*)d_in[5];
  const float* rel_pos_w = (const float*)d_in[6];
  // d_in[7], d_in[8] are H, W device scalars; fixed at 128 per setup_inputs.

  const int nTok = in_sizes[0] / 256;            // B * 128 * 128 = 262144
  const size_t perBuf = (size_t)nTok * 256;      // f16 elements each

  _Float16* qs  = (_Float16*)d_ws;
  _Float16* kk  = qs + perBuf;
  _Float16* vv  = kk + perBuf;
  _Float16* yw  = vv + perBuf;
  _Float16* xh  = yw + perBuf;
  _Float16* qwh = xh + perBuf;
  _Float16* pwh = qwh + 768 * 256;
  float* out = (float*)d_out;

  const long long nx = (long long)nTok * 256;
  const long long totalCvt = nx + 768LL * 256 + 256LL * 256;
  cvt_kernel<<<dim3((unsigned)(totalCvt / 2048)), 256, 0, stream>>>(
      x, qkv_w, proj_w, xh, qwh, pwh, nx);

  const int mTiles = nTok / 128;                 // 2048
  // grid.x = N tiles so consecutive blocks share the A tile (L2 reuse)
  qkv_kernel<<<dim3(12, mTiles), 256, 0, stream>>>(xh, qwh, qkv_b, qs, kk, vv);

  const int nWinHeads = (nTok / 64) * 8;         // 32768 window-head problems
  attn_kernel<<<dim3(nWinHeads), 128, 0, stream>>>(qs, kk, vv, rel_pos_h, rel_pos_w, yw);

  proj_kernel<<<dim3(4, mTiles), 256, 0, stream>>>(yw, pwh, proj_b, out);
}
